// ScaledDotProductAttention_17721035064078
// MI455X (gfx1250) — matrix operand order repaired and checked by reading, not compiled, not run on hardware
//
#include <hip/hip_runtime.h>

// ---------------------------------------------------------------------------
// Problem constants (B,S,D,H from the reference)
// ---------------------------------------------------------------------------
constexpr int Bb  = 4;
constexpr int Ss  = 2048;
constexpr int Dd  = 1024;
constexpr int Hh  = 16;
constexpr int DK  = 64;
constexpr int MT  = Bb * Ss;          // 8192 total tokens

typedef __attribute__((ext_vector_type(16))) __bf16 v16bf;
typedef __attribute__((ext_vector_type(8)))  float  v8f;

__device__ inline v8f wmma_bf16(v16bf a, v16bf b, v8f c) {
    return __builtin_amdgcn_wmma_f32_16x16x32_bf16(
        /*neg_a=*/false, a, /*neg_b=*/false, b,
        /*c_mod=*/(short)0, c, /*reuse_a=*/false, /*reuse_b=*/false);
}

// Async copy of one 16B chunk: global -> LDS (ASYNCcnt-tracked).
__device__ inline void async_cp16(const void* gptr, const void* lptr) {
    unsigned long long ga = (unsigned long long)(uintptr_t)gptr;
    unsigned           la = (unsigned)(uintptr_t)lptr;  // low 32 bits = LDS offset
    asm volatile("global_load_async_to_lds_b128 %0, %1, off"
                 :: "v"(la), "v"(ga) : "memory");
}
__device__ inline void wait_async0()  { asm volatile("s_wait_asynccnt 0x0" ::: "memory"); }
__device__ inline void wait_async16() { asm volatile("s_wait_asynccnt 0x10" ::: "memory"); }

// ---------------------------------------------------------------------------
// Fragment loaders (ISA 7.12.2 layouts), b128 form.
// A 16x32 bf16: lane l -> row m=l&15, half h=l>>4.  The lane's 16 elements
//   are K = [8h, 8h+8) then [16+8h, 16+8h+8): two 16B runs.
// B 32x16 bf16: lane l -> col n=l&15; elements are K = [8h, 8h+8) then [16+8h, 16+8h+8): two 16B runs.
// rowPtr must be 16B aligned.
// ---------------------------------------------------------------------------
__device__ inline v16bf frag_a(const __bf16* rowPtr, int h) {
    union { uint4 u[2]; v16bf v; } cv;
    cv.u[0] = *(const uint4*)(rowPtr + 8 * h);
    cv.u[1] = *(const uint4*)(rowPtr + 16 + 8 * h);
    return cv.v;
}

__device__ inline v16bf frag_b(const __bf16* rowPtr, int h) {
    union { uint4 u[2]; v16bf v; } cv;
    cv.u[0] = *(const uint4*)(rowPtr + 8 * h);
    cv.u[1] = *(const uint4*)(rowPtr + 16 + 8 * h);
    return cv.v;
}

__device__ inline v8f vzero8() {
    v8f z;
#pragma unroll
    for (int i = 0; i < 8; ++i) z[i] = 0.0f;
    return z;
}

// ---------------------------------------------------------------------------
// One-time fp32 -> bf16 conversion (8 elements / thread).
// ---------------------------------------------------------------------------
__global__ void cvt_kernel(const float* __restrict__ in,
                           __bf16* __restrict__ out, int n) {
    int i = (blockIdx.x * blockDim.x + threadIdx.x) * 8;
    if (i >= n) return;
    float4 a = *(const float4*)(in + i);
    float4 b = *(const float4*)(in + i + 4);
    union { __bf16 h[8]; uint4 u; } pk;
    pk.h[0] = (__bf16)a.x; pk.h[1] = (__bf16)a.y;
    pk.h[2] = (__bf16)a.z; pk.h[3] = (__bf16)a.w;
    pk.h[4] = (__bf16)b.x; pk.h[5] = (__bf16)b.y;
    pk.h[6] = (__bf16)b.z; pk.h[7] = (__bf16)b.w;
    *(uint4*)(out + i) = pk.u;
}

// ---------------------------------------------------------------------------
// Projection GEMM on bf16 inputs:  Y[m][n] = sum_k X[m][k] * W[n][k]
// Block tile 128x128, 8 waves, wave tile 32(M) x 64(N), K-step 64.
// Double-buffered async LDS staging (bf16 tiles, 72-elem padded rows).
// mode 0: bf16 row-major [MT][Dd] (Q scaled 1/8, K).  mode 1: [Dd][MT] (V^T).
// ---------------------------------------------------------------------------
__global__ void proj_kernel(const __bf16* __restrict__ X,
                            const __bf16* __restrict__ W,
                            __bf16* __restrict__ Y,
                            int transposed, float scale) {
    __shared__ __bf16 Xs[2][128][72];   // 64 used + 8 pad (144B rows, 16B align)
    __shared__ __bf16 Ws[2][128][72];

    const int tid  = threadIdx.x;
    const int w    = tid >> 5;
    const int lane = tid & 31;
    const int l15  = lane & 15;
    const int hf   = lane >> 4;

    const int m0b = blockIdx.x * 128;
    const int n0b = blockIdx.y * 128;
    const int m0w = (w >> 1) * 32;      // wave M offset within block
    const int n0w = (w & 1) * 64;       // wave N offset within block

    v8f acc[2][4];
#pragma unroll
    for (int i = 0; i < 2; ++i)
#pragma unroll
        for (int j = 0; j < 4; ++j) acc[i][j] = vzero8();

    // stage one 128x64 bf16 tile pair into LDS buffer `buf` for k0
    auto fill = [&](int buf, int k0) {
#pragma unroll
        for (int i = 0; i < 4; ++i) {                 // 1024 chunks / 256 thr
            int c   = tid + i * 256;
            int row = c >> 3;                         // 8 x 16B per 128B row
            int col = (c & 7) * 8;                    // bf16 units
            async_cp16(X + (size_t)(m0b + row) * Dd + k0 + col,
                       &Xs[buf][row][col]);
            async_cp16(W + (size_t)(n0b + row) * Dd + k0 + col,
                       &Ws[buf][row][col]);
        }
    };

    fill(0, 0);
    constexpr int NSTEP = Dd / 64;                    // 16
    for (int s = 0; s < NSTEP; ++s) {
        const int buf = s & 1;
        wait_async0();
        __syncthreads();
        if (s + 1 < NSTEP) fill(buf ^ 1, (s + 1) * 64);

#pragma unroll
        for (int kk = 0; kk < 64; kk += 32) {
            v16bf a0 = frag_a(&Xs[buf][m0w + l15][kk], hf);
            v16bf a1 = frag_a(&Xs[buf][m0w + 16 + l15][kk], hf);
            v16bf b0 = frag_b(&Ws[buf][n0w + l15][kk], hf);
            v16bf b1 = frag_b(&Ws[buf][n0w + 16 + l15][kk], hf);
            v16bf b2 = frag_b(&Ws[buf][n0w + 32 + l15][kk], hf);
            v16bf b3 = frag_b(&Ws[buf][n0w + 48 + l15][kk], hf);
            acc[0][0] = wmma_bf16(a0, b0, acc[0][0]);
            acc[0][1] = wmma_bf16(a0, b1, acc[0][1]);
            acc[0][2] = wmma_bf16(a0, b2, acc[0][2]);
            acc[0][3] = wmma_bf16(a0, b3, acc[0][3]);
            acc[1][0] = wmma_bf16(a1, b0, acc[1][0]);
            acc[1][1] = wmma_bf16(a1, b1, acc[1][1]);
            acc[1][2] = wmma_bf16(a1, b2, acc[1][2]);
            acc[1][3] = wmma_bf16(a1, b3, acc[1][3]);
        }
    }

#pragma unroll
    for (int i = 0; i < 2; ++i) {
#pragma unroll
        for (int j = 0; j < 4; ++j) {
            const int mb = m0b + m0w + i * 16;
            const int nb = n0b + n0w + j * 16;
            if (!transposed) {
#pragma unroll
                for (int r = 0; r < 8; ++r) {
                    int m = mb + r + 8 * hf;
                    int n = nb + l15;
                    Y[(size_t)m * Dd + n] = (__bf16)(acc[i][j][r] * scale);
                }
            } else {
                // lane holds 8 consecutive tokens (mb+8h..+7) of column n
                int n = nb + l15;
                union { __bf16 b[8]; uint4 u; } pk;
#pragma unroll
                for (int r = 0; r < 8; ++r) pk.b[r] = (__bf16)(acc[i][j][r] * scale);
                *(uint4*)(Y + (size_t)n * MT + mb + 8 * hf) = pk.u;
            }
        }
    }
}

// ---------------------------------------------------------------------------
// Flash attention: 4 waves/block, one wave = 16 query rows of one (b, head).
// Per-wave double-buffered async staging of 32-key K/V chunks (no barriers:
// ASYNCcnt is per-wave; partial wait `asynccnt<=16` overlaps next-chunk DMA).
// Qbf/Kbf bf16 [MT][Dd] (Q pre-scaled 1/8); Vt bf16 [Dd][MT]; Out f32.
// ---------------------------------------------------------------------------
__global__ void attn_kernel(const __bf16* __restrict__ Qbf,
                            const __bf16* __restrict__ Kbf,
                            const __bf16* __restrict__ Vt,
                            float* __restrict__ Out) {
    __shared__ __bf16 Ks[4][2][32][72];   // 64 dk used + 8 pad (144B rows)
    __shared__ __bf16 Vs[4][2][64][40];   // 32 keys used + 8 pad (80B rows)
    __shared__ __bf16 Plds[4][16][40];    // per-wave P tile (80B rows)

    const int w    = threadIdx.x >> 5;
    const int lane = threadIdx.x & 31;
    const int l15  = lane & 15;
    const int hf   = lane >> 4;

    const int nTiles = Ss / 64;                     // 32 (4 waves x 16 rows)
    const int tile   = blockIdx.x % nTiles;
    const int bh     = blockIdx.x / nTiles;
    const int b      = bh / Hh;
    const int head   = bh % Hh;
    const int q0     = tile * 64 + w * 16;
    const int bS     = b * Ss;

    // stage a 32-key chunk: K rows (32 x 128B) + V^T rows (64 x 64B) = 16 ops
    auto fillKV = [&](int buf, int k0) {
#pragma unroll
        for (int i = 0; i < 8; ++i) {
            int c = i * 32 + lane;
            {   // K: 8 x16B chunks per key row
                int row = c >> 3, col = (c & 7) * 8;
                async_cp16(Kbf + (size_t)(bS + k0 + row) * Dd + head * DK + col,
                           &Ks[w][buf][row][col]);
            }
            {   // V^T: 4 x16B chunks per dk row
                int row = c >> 2, col = (c & 3) * 8;
                async_cp16(Vt + (size_t)(head * DK + row) * MT + bS + k0 + col,
                           &Vs[w][buf][row][col]);
            }
        }
    };

    // Q fragments for this wave's 16 rows (DK=64 -> two K=32 A-frags)
    const __bf16* qrow = Qbf + (size_t)(bS + q0 + l15) * Dd + head * DK;
    v16bf qf0 = frag_a(qrow, hf);
    v16bf qf1 = frag_a(qrow + 32, hf);

    v8f out[4];
#pragma unroll
    for (int f = 0; f < 4; ++f) out[f] = vzero8();
    float mrow[8], lrow[8];
#pragma unroll
    for (int r = 0; r < 8; ++r) { mrow[r] = -1e30f; lrow[r] = 0.0f; }

    const int kend = q0 + 16;                       // causal upper bound
    const int nCh  = (kend + 31) / 32;

    fillKV(0, 0);
    for (int c = 0; c < nCh; ++c) {
        const int buf = c & 1;
        const int k0  = c * 32;
        if (c + 1 < nCh) { fillKV(buf ^ 1, k0 + 32); wait_async16(); }
        else             { wait_async0(); }

        // ---- scores: S(16x32) = Q(16x64) * K^T  (4 WMMAs) ----
        v8f sA = vzero8(), sB = vzero8();
        {
            const __bf16* kr0 = &Ks[w][buf][l15][0];
            const __bf16* kr1 = &Ks[w][buf][16 + l15][0];
            sA = wmma_bf16(qf0, frag_b(kr0, hf), sA);
            sB = wmma_bf16(qf0, frag_b(kr1, hf), sB);
            sA = wmma_bf16(qf1, frag_b(kr0 + 32, hf), sA);
            sB = wmma_bf16(qf1, frag_b(kr1 + 32, hf), sB);
        }

        const bool doMask = (k0 + 31) > q0;

        // ---- online softmax over the 32-key chunk ----
#pragma unroll
        for (int r = 0; r < 8; ++r) {
            const int q = q0 + r + 8 * hf;
            float s0 = sA[r];
            float s1 = sB[r];
            if (doMask) {
                if (k0 + l15 > q)      s0 = -1e30f;
                if (k0 + 16 + l15 > q) s1 = -1e30f;
            }
            float mx = fmaxf(s0, s1);
            mx = fmaxf(mx, __shfl_xor(mx, 1, 16));
            mx = fmaxf(mx, __shfl_xor(mx, 2, 16));
            mx = fmaxf(mx, __shfl_xor(mx, 4, 16));
            mx = fmaxf(mx, __shfl_xor(mx, 8, 16));
            const float mnew  = fmaxf(mrow[r], mx);
            const float p0    = __expf(s0 - mnew);
            const float p1    = __expf(s1 - mnew);
            const float alpha = __expf(mrow[r] - mnew);
            mrow[r] = mnew;
            float rs = p0 + p1;
            rs += __shfl_xor(rs, 1, 16);
            rs += __shfl_xor(rs, 2, 16);
            rs += __shfl_xor(rs, 4, 16);
            rs += __shfl_xor(rs, 8, 16);
            lrow[r] = lrow[r] * alpha + rs;
#pragma unroll
            for (int f = 0; f < 4; ++f) out[f][r] *= alpha;
            Plds[w][r + 8 * hf][l15]      = (__bf16)p0;
            Plds[w][r + 8 * hf][16 + l15] = (__bf16)p1;
        }

        // per-wave LDS store -> load ordering
        asm volatile("s_wait_dscnt 0x0" ::: "memory");

        // ---- P(16x32) -> A-fragment, then O += P * V (4 WMMAs) ----
        v16bf pa = frag_a(&Plds[w][l15][0], hf);
#pragma unroll
        for (int f = 0; f < 4; ++f) {
            v16bf bv = frag_b(&Vs[w][buf][f * 16 + l15][0], hf);
            out[f] = wmma_bf16(pa, bv, out[f]);
        }
    }

    // ---- normalize and store ----
#pragma unroll
    for (int r = 0; r < 8; ++r) {
        const float inv = 1.0f / lrow[r];
        const int q = q0 + r + 8 * hf;
#pragma unroll
        for (int f = 0; f < 4; ++f) {
            const int col = head * DK + f * 16 + l15;
            Out[(size_t)(bS + q) * Dd + col] = out[f][r] * inv;
        }
    }
}

// ---------------------------------------------------------------------------
// Launch
// ---------------------------------------------------------------------------
extern "C" void kernel_launch(void* const* d_in, const int* in_sizes, int n_in,
                              void* d_out, int out_size, void* d_ws, size_t ws_size,
                              hipStream_t stream) {
    (void)in_sizes; (void)n_in; (void)out_size; (void)ws_size;
    const float* X  = (const float*)d_in[0];
    const float* Wq = (const float*)d_in[1];
    const float* Wk = (const float*)d_in[2];
    const float* Wv = (const float*)d_in[3];

    __bf16* Qbf = (__bf16*)d_ws;                       // [MT][Dd]
    __bf16* Kbf = Qbf + (size_t)MT * Dd;               // [MT][Dd]
    __bf16* Vt  = Kbf + (size_t)MT * Dd;               // [Dd][MT]
    __bf16* Xb  = Vt  + (size_t)MT * Dd;               // [MT][Dd]  bf16 inputs
    __bf16* Wqb = Xb  + (size_t)MT * Dd;               // [Dd][Dd]
    __bf16* Wkb = Wqb + (size_t)Dd * Dd;
    __bf16* Wvb = Wkb + (size_t)Dd * Dd;

    // one-time bf16 conversion of X and the three weight matrices
    {
        int nX = MT * Dd, nW = Dd * Dd;
        cvt_kernel<<<nX / 8 / 256, 256, 0, stream>>>(X,  Xb,  nX);
        cvt_kernel<<<nW / 8 / 256, 256, 0, stream>>>(Wq, Wqb, nW);
        cvt_kernel<<<nW / 8 / 256, 256, 0, stream>>>(Wk, Wkb, nW);
        cvt_kernel<<<nW / 8 / 256, 256, 0, stream>>>(Wv, Wvb, nW);
    }

    dim3 gridP(MT / 128, Dd / 128);
    const float qscale = 0.125f;                       // 1/sqrt(DK)
    proj_kernel<<<gridP, 256, 0, stream>>>(Xb, Wqb, Qbf, 0, qscale);
    proj_kernel<<<gridP, 256, 0, stream>>>(Xb, Wkb, Kbf, 0, 1.0f);
    proj_kernel<<<gridP, 256, 0, stream>>>(Xb, Wvb, Vt,  1, 1.0f);

    dim3 gridA(Bb * Hh * (Ss / 64));
    attn_kernel<<<gridA, 128, 0, stream>>>(Qbf, Kbf, Vt, (float*)d_out);
}
